// SlotAttention_90726889160915
// MI455X (gfx1250) — compile-verified, matrix operand order checked by reading, not run on hardware
//
#include <hip/hip_runtime.h>
#include <math.h>

#define B_SZ   32
#define N_SZ   16384
#define K_SL   8
#define D_DIM  64
#define H_DIM  128
#define T_IT   3
#define EPS_C  1e-8f
#define LNEPS  1e-5f

typedef __attribute__((ext_vector_type(16))) __bf16          v16bf_t;
typedef __attribute__((ext_vector_type(8)))  float           v8f_t;
typedef __attribute__((ext_vector_type(8)))  unsigned short  ushort8_t;

// ---------- helpers ----------
__device__ __forceinline__ unsigned short f2bf(float f) {
  unsigned u = __float_as_uint(f);
  u += 0x7fffu + ((u >> 16) & 1u);          // round-to-nearest-even
  return (unsigned short)(u >> 16);
}

// Load one 16x32 bf16 WMMA operand fragment (A or B) from a row-major
// matrix: lane&15 selects the row (A: M row, B: N column of the stored
// operand matrix), contracted dim is contiguous in memory.
// Per ISA 7.12.2: lanes 0-15 hold K = kb+{0..7, 16..23}, lanes 16-31 hold
// K = kb+{8..15, 24..31}.
__device__ __forceinline__ v16bf_t load_frag(const unsigned short* base,
                                             int ld, int r0, int kb) {
  const int lane = threadIdx.x & 31;
  const int l15 = lane & 15, hl = lane >> 4;
  const unsigned short* p = base + (size_t)(r0 + l15) * ld + kb + hl * 8;
  union { v16bf_t v; ushort8_t h[2]; } u;
  u.h[0] = *reinterpret_cast<const ushort8_t*>(p);
  u.h[1] = *reinterpret_cast<const ushort8_t*>(p + 16);
  return u.v;
}

__device__ __forceinline__ v8f_t wmma_bf16(v16bf_t a, v16bf_t b, v8f_t c) {
  return __builtin_amdgcn_wmma_f32_16x16x32_bf16(false, a, false, b,
                                                 (short)0, c, false, false);
}

// ---------- kernel 0: fp32 -> bf16 weight convert ----------
__global__ void k_cvt_bf16(const float* __restrict__ src,
                           unsigned short* __restrict__ dst, int n) {
  int i = blockIdx.x * blockDim.x + threadIdx.x;
  if (i < n) dst[i] = f2bf(src[i]);
}

// ---------- kernel 1: LN(inputs) then k = x@Wk^T, v = x@Wv^T ----------
// One block = 64 input rows (one batch slice). k stored row-major bf16
// (B,N,64); v stored transposed bf16 (B,64,N) so the updates-GEMM B-frags
// are contiguous.
__global__ __launch_bounds__(256) void k_lnproj(
    const float* __restrict__ inp, const float* __restrict__ g_in,
    const float* __restrict__ b_in, const unsigned short* __restrict__ Wk_bf,
    const unsigned short* __restrict__ Wv_bf,
    unsigned short* __restrict__ k_g, unsigned short* __restrict__ vT_g) {
  __shared__ __align__(16) unsigned short xb[64][64];   // LN'd inputs, bf16
  __shared__ __align__(16) unsigned short ktr[64][64];  // k transpose staging

  const int t = threadIdx.x;
  const size_t R0 = (size_t)blockIdx.x * 64;
  const int b = (int)(R0 / N_SZ);
  const int n_base = (int)(R0 % N_SZ);

  // ---- LayerNorm: 4 threads per row, 16 elements each ----
  {
    const int row = t >> 2, part = t & 3;
    const float* src = inp + ((size_t)b * N_SZ + n_base + row) * D_DIM + part * 16;
    float x[16];
    float s = 0.f, q = 0.f;
#pragma unroll
    for (int j = 0; j < 16; ++j) { x[j] = src[j]; s += x[j]; q += x[j] * x[j]; }
    s += __shfl_xor(s, 1); s += __shfl_xor(s, 2);
    q += __shfl_xor(q, 1); q += __shfl_xor(q, 2);
    const float mean = s * (1.f / 64.f);
    const float var = q * (1.f / 64.f) - mean * mean;
    const float rstd = rsqrtf(var + LNEPS);
#pragma unroll
    for (int j = 0; j < 16; ++j) {
      const int e = part * 16 + j;
      xb[row][e] = f2bf((x[j] - mean) * rstd * g_in[e] + b_in[e]);
    }
  }
  __syncthreads();

  // ---- WMMA: wave w -> projection (w&1), row tile (w>>1) ----
  const int wave = t >> 5;
  const int proj = wave & 1;          // 0 = k, 1 = v
  const int rt = wave >> 1;           // 16-row tile
  const int lane = t & 31, l15 = lane & 15, hl = lane >> 4;

  const v16bf_t a0 = load_frag(&xb[0][0], 64, rt * 16, 0);
  const v16bf_t a1 = load_frag(&xb[0][0], 64, rt * 16, 32);
  const unsigned short* W = proj ? Wv_bf : Wk_bf;

#pragma unroll
  for (int ct = 0; ct < 4; ++ct) {
    v8f_t acc = {0.f, 0.f, 0.f, 0.f, 0.f, 0.f, 0.f, 0.f};
    acc = wmma_bf16(a0, load_frag(W, 64, ct * 16, 0), acc);
    acc = wmma_bf16(a1, load_frag(W, 64, ct * 16, 32), acc);
    const int d = ct * 16 + l15;              // output column held by lane
    if (proj == 0) {                          // k: stage transpose in LDS
#pragma unroll
      for (int g = 0; g < 8; ++g) {
        const int m = g + 8 * hl;             // row within tile
        ktr[rt * 16 + m][d] = f2bf(acc[g]);
      }
    } else {                                  // v: direct contiguous vT store
      ushort8_t pk;
#pragma unroll
      for (int g = 0; g < 8; ++g) pk[g] = f2bf(acc[g]);
      const int n0 = n_base + rt * 16 + hl * 8;
      *reinterpret_cast<ushort8_t*>(
          vT_g + ((size_t)b * D_DIM + d) * N_SZ + n0) = pk;
    }
  }
  __syncthreads();

  // ---- contiguous k copy-out ----
  {
    const int row = t >> 2, part = t & 3;
    unsigned short* dst =
        k_g + ((size_t)b * N_SZ + n_base + row) * D_DIM + part * 16;
    *reinterpret_cast<ushort8_t*>(dst) =
        *reinterpret_cast<const ushort8_t*>(&ktr[row][part * 16]);
    *reinterpret_cast<ushort8_t*>(dst + 8) =
        *reinterpret_cast<const ushort8_t*>(&ktr[row][part * 16 + 8]);
  }
}

// ---------- kernel 2: q = (LN(slots) @ Wq^T) * scale, bf16, (B,16,64) ----------
__global__ __launch_bounds__(32) void k_qproj(
    const float* __restrict__ slots, const float* __restrict__ g_s,
    const float* __restrict__ b_s, const unsigned short* __restrict__ Wq_bf,
    unsigned short* __restrict__ q_g) {
  __shared__ __align__(16) unsigned short sb[16][64];
  __shared__ __align__(16) unsigned short qtr[16][64];
  const int b = blockIdx.x;
  const int lane = threadIdx.x & 31, l15 = lane & 15, hl = lane >> 4;

  if (lane < K_SL) {
    const float* sr = slots + ((size_t)b * K_SL + lane) * D_DIM;
    float s = 0.f, q = 0.f;
    for (int j = 0; j < 64; ++j) { float x = sr[j]; s += x; q += x * x; }
    const float mean = s * (1.f / 64.f);
    const float rstd = rsqrtf(q * (1.f / 64.f) - mean * mean + LNEPS);
    for (int j = 0; j < 64; ++j)
      sb[lane][j] = f2bf((sr[j] - mean) * rstd * g_s[j] + b_s[j]);
  } else if (lane < 16) {
    for (int j = 0; j < 64; ++j) sb[lane][j] = 0;  // pad rows 8..15
  }
  __syncthreads();

  const v16bf_t a0 = load_frag(&sb[0][0], 64, 0, 0);
  const v16bf_t a1 = load_frag(&sb[0][0], 64, 0, 32);
#pragma unroll
  for (int ct = 0; ct < 4; ++ct) {
    v8f_t acc = {0.f, 0.f, 0.f, 0.f, 0.f, 0.f, 0.f, 0.f};
    acc = wmma_bf16(a0, load_frag(Wq_bf, 64, ct * 16, 0), acc);
    acc = wmma_bf16(a1, load_frag(Wq_bf, 64, ct * 16, 32), acc);
#pragma unroll
    for (int g = 0; g < 8; ++g)
      qtr[g + 8 * hl][ct * 16 + l15] = f2bf(acc[g] * 0.125f);  // fold 1/sqrt(D)
  }
  __syncthreads();

  const int row = lane >> 1, half = lane & 1;
  unsigned short* dst = q_g + (size_t)b * 16 * 64 + row * 64 + half * 32;
#pragma unroll
  for (int g = 0; g < 4; ++g)
    *reinterpret_cast<ushort8_t*>(dst + g * 8) =
        *reinterpret_cast<const ushort8_t*>(&qtr[row][half * 32 + g * 8]);
}

// ---------- kernel 3: logits = q @ k^T  -> (B,8,N) fp32 ----------
__global__ __launch_bounds__(256) void k_logits(
    const unsigned short* __restrict__ q_g,
    const unsigned short* __restrict__ k_g, float* __restrict__ Lg) {
  const int b = blockIdx.y;
  const int wave = threadIdx.x >> 5;
  const int n0 = blockIdx.x * 128 + wave * 16;
  const int lane = threadIdx.x & 31, l15 = lane & 15, hl = lane >> 4;

  const unsigned short* qb = q_g + (size_t)b * 16 * 64;
  const unsigned short* kb = k_g + (size_t)b * N_SZ * 64;

  v8f_t acc = {0.f, 0.f, 0.f, 0.f, 0.f, 0.f, 0.f, 0.f};
  acc = wmma_bf16(load_frag(qb, 64, 0, 0),  load_frag(kb, 64, n0, 0),  acc);
  acc = wmma_bf16(load_frag(qb, 64, 0, 32), load_frag(kb, 64, n0, 32), acc);

  if (hl == 0) {                        // rows 0..7 = valid slots
#pragma unroll
    for (int g = 0; g < 8; ++g)
      Lg[((size_t)b * K_SL + g) * N_SZ + n0 + l15] = acc[g];
  }
}

// ---------- kernel 4: per-row online softmax stats (max, sumexp) ----------
__global__ __launch_bounds__(256) void k_stats(const float* __restrict__ Lg,
                                               float* __restrict__ stats) {
  __shared__ float sm[256], ss[256];
  const int row = blockIdx.x;           // b*8 + slot
  const int t = threadIdx.x;
  float m = -3.4e38f, s = 0.f;
  const float* src = Lg + (size_t)row * N_SZ;
  for (int j = t; j < N_SZ; j += 256) {
    const float x = src[j];
    const float nm = fmaxf(m, x);
    s = s * __expf(m - nm) + __expf(x - nm);
    m = nm;
  }
  sm[t] = m; ss[t] = s;
  __syncthreads();
  for (int off = 128; off > 0; off >>= 1) {
    if (t < off) {
      const float m1 = sm[t], s1 = ss[t], m2 = sm[t + off], s2 = ss[t + off];
      const float nm = fmaxf(m1, m2);
      sm[t] = nm;
      ss[t] = s1 * __expf(m1 - nm) + s2 * __expf(m2 - nm);
    }
    __syncthreads();
  }
  if (t == 0) { stats[2 * row] = sm[0]; stats[2 * row + 1] = ss[0]; }
}

// ---------- kernel 5: attn = softmax + EPS, slot competition ----------
__global__ __launch_bounds__(256) void k_attn(
    const float* __restrict__ Lg, const float* __restrict__ stats,
    unsigned short* __restrict__ attnb, float* __restrict__ updates,
    float* __restrict__ attn_out, int last) {
  const int b = blockIdx.y;
  const int n = blockIdx.x * 256 + threadIdx.x;

  if (blockIdx.x == 0) {                // zero the fp32 update accumulators
    updates[(size_t)b * 512 + threadIdx.x] = 0.f;
    updates[(size_t)b * 512 + 256 + threadIdx.x] = 0.f;
  }

  float p[K_SL];
  float den = 0.f;
#pragma unroll
  for (int i = 0; i < K_SL; ++i) {
    const int row = b * K_SL + i;
    const float m = stats[2 * row];
    const float inv_s = 1.f / stats[2 * row + 1];
    p[i] = __expf(Lg[(size_t)row * N_SZ + n] - m) * inv_s + EPS_C;
    den += p[i];
  }
  const float inv_den = 1.f / den;
#pragma unroll
  for (int i = 0; i < K_SL; ++i) {
    const float a = p[i] * inv_den;
    attnb[((size_t)b * 16 + i) * N_SZ + n] = f2bf(a);
    if (last) attn_out[((size_t)b * K_SL + i) * N_SZ + n] = a;
  }
}

// ---------- kernel 6: updates = attn @ v  (WMMA over n, fp32 atomics) ----------
__global__ __launch_bounds__(128) void k_updates(
    const unsigned short* __restrict__ attnb,
    const unsigned short* __restrict__ vT, float* __restrict__ updates) {
  const int b = blockIdx.y;
  const int nb0 = blockIdx.x * 512;
  const int ct = threadIdx.x >> 5;            // column tile of D (4 waves)
  const int lane = threadIdx.x & 31, l15 = lane & 15, hl = lane >> 4;

  const unsigned short* ab = attnb + (size_t)b * 16 * N_SZ;
  const unsigned short* vb = vT + (size_t)b * D_DIM * N_SZ;

  v8f_t acc = {0.f, 0.f, 0.f, 0.f, 0.f, 0.f, 0.f, 0.f};
#pragma unroll 4
  for (int s = 0; s < 16; ++s) {
    const int kb = nb0 + s * 32;
    acc = wmma_bf16(load_frag(ab, N_SZ, 0, kb),
                    load_frag(vb, N_SZ, ct * 16, kb), acc);
  }
  if (hl == 0) {                              // rows 0..7 = valid slots
    const int d = ct * 16 + l15;
#pragma unroll
    for (int g = 0; g < 8; ++g)
      atomicAdd(&updates[(size_t)b * 512 + g * 64 + d], acc[g]);
  }
}

// ---------- kernel 7: GRU cell + LN + MLP residual (fp32 VALU) ----------
__global__ __launch_bounds__(64) void k_gru(
    const float* __restrict__ updates, float* __restrict__ slots,
    const float* __restrict__ W_ih, const float* __restrict__ W_hh,
    const float* __restrict__ b_ih, const float* __restrict__ b_hh,
    const float* __restrict__ g_m, const float* __restrict__ b_m,
    const float* __restrict__ W1, const float* __restrict__ b1,
    const float* __restrict__ W2, const float* __restrict__ b2,
    float* __restrict__ out_slots, int last) {
  __shared__ float xs[64], hs[64], red[64], lnb[64], hid[128];
  const int row = blockIdx.x;                 // b*8 + slot
  const int t = threadIdx.x;

  const float x_t = updates[(size_t)row * 64 + t];
  const float h_t = slots[(size_t)row * 64 + t];
  xs[t] = x_t; hs[t] = h_t;
  __syncthreads();

  float xr = b_ih[t], xz = b_ih[64 + t], xn = b_ih[128 + t];
  float hr = b_hh[t], hz = b_hh[64 + t], hn = b_hh[128 + t];
  for (int j = 0; j < 64; ++j) {
    const float xv = xs[j], hv = hs[j];
    xr += xv * W_ih[(size_t)t * 64 + j];
    xz += xv * W_ih[(size_t)(64 + t) * 64 + j];
    xn += xv * W_ih[(size_t)(128 + t) * 64 + j];
    hr += hv * W_hh[(size_t)t * 64 + j];
    hz += hv * W_hh[(size_t)(64 + t) * 64 + j];
    hn += hv * W_hh[(size_t)(128 + t) * 64 + j];
  }
  const float r = 1.f / (1.f + __expf(-(xr + hr)));
  const float z = 1.f / (1.f + __expf(-(xz + hz)));
  const float ng = tanhf(xn + r * hn);
  const float s = (1.f - z) * ng + z * h_t;

  // LayerNorm over the 64 slot features
  red[t] = s; __syncthreads();
  for (int off = 32; off > 0; off >>= 1) {
    if (t < off) red[t] += red[t + off];
    __syncthreads();
  }
  const float mean = red[0] * (1.f / 64.f);
  __syncthreads();
  const float dv = s - mean;
  red[t] = dv * dv; __syncthreads();
  for (int off = 32; off > 0; off >>= 1) {
    if (t < off) red[t] += red[t + off];
    __syncthreads();
  }
  const float rstd = rsqrtf(red[0] * (1.f / 64.f) + LNEPS);
  lnb[t] = dv * rstd * g_m[t] + b_m[t];
  __syncthreads();

  // MLP hidden: each thread computes 2 of 128 hidden units
#pragma unroll
  for (int hh = 0; hh < 2; ++hh) {
    const int hrow = t + hh * 64;
    float a = b1[hrow];
    for (int j = 0; j < 64; ++j) a += lnb[j] * W1[(size_t)hrow * 64 + j];
    hid[hrow] = fmaxf(a, 0.f);
  }
  __syncthreads();

  float o = s + b2[t];
  for (int j = 0; j < 128; ++j) o += hid[j] * W2[(size_t)t * 128 + j];

  slots[(size_t)row * 64 + t] = o;
  if (last) out_slots[(size_t)row * 64 + t] = o;
}

// ---------- host launch ----------
extern "C" void kernel_launch(void* const* d_in, const int* in_sizes, int n_in,
                              void* d_out, int out_size, void* d_ws,
                              size_t ws_size, hipStream_t stream) {
  (void)in_sizes; (void)n_in; (void)out_size; (void)ws_size;
  const float* inputs     = (const float*)d_in[0];
  const float* slots_init = (const float*)d_in[1];
  const float* g_in = (const float*)d_in[2];
  const float* b_in = (const float*)d_in[3];
  const float* g_s  = (const float*)d_in[4];
  const float* b_s  = (const float*)d_in[5];
  const float* g_m  = (const float*)d_in[6];
  const float* b_m  = (const float*)d_in[7];
  const float* Wq   = (const float*)d_in[8];
  const float* Wk   = (const float*)d_in[9];
  const float* Wv   = (const float*)d_in[10];
  const float* W_ih = (const float*)d_in[11];
  const float* W_hh = (const float*)d_in[12];
  const float* b_ih = (const float*)d_in[13];
  const float* b_hh = (const float*)d_in[14];
  const float* W1   = (const float*)d_in[15];
  const float* b1   = (const float*)d_in[16];
  const float* W2   = (const float*)d_in[17];
  const float* b2   = (const float*)d_in[18];

  float* out = (float*)d_out;
  float* out_slots = out;                                   // (B,8,64)
  float* out_attn  = out + (size_t)B_SZ * K_SL * D_DIM;     // (B,8,N)

  // workspace carve-up
  char* w = (char*)d_ws;
  auto take = [&](size_t bytes) {
    char* p = w;
    w += (bytes + 255) & ~(size_t)255;
    return p;
  };
  unsigned short* k_g   = (unsigned short*)take((size_t)B_SZ * N_SZ * D_DIM * 2);
  unsigned short* vT    = (unsigned short*)take((size_t)B_SZ * D_DIM * N_SZ * 2);
  unsigned short* attnb = (unsigned short*)take((size_t)B_SZ * 16 * N_SZ * 2);
  float*          Lg    = (float*)take((size_t)B_SZ * K_SL * N_SZ * 4);
  unsigned short* q_g   = (unsigned short*)take((size_t)B_SZ * 16 * D_DIM * 2);
  unsigned short* Wq_bf = (unsigned short*)take(4096 * 2);
  unsigned short* Wk_bf = (unsigned short*)take(4096 * 2);
  unsigned short* Wv_bf = (unsigned short*)take(4096 * 2);
  float*          stats = (float*)take(2 * B_SZ * K_SL * 4);
  float*          updts = (float*)take((size_t)B_SZ * K_SL * D_DIM * 4);
  float*          slotb = (float*)take((size_t)B_SZ * K_SL * D_DIM * 4);

  // weight conversion + slots init
  k_cvt_bf16<<<16, 256, 0, stream>>>(Wq, Wq_bf, 4096);
  k_cvt_bf16<<<16, 256, 0, stream>>>(Wk, Wk_bf, 4096);
  k_cvt_bf16<<<16, 256, 0, stream>>>(Wv, Wv_bf, 4096);
  hipMemcpyAsync(slotb, slots_init, (size_t)B_SZ * K_SL * D_DIM * 4,
                 hipMemcpyDeviceToDevice, stream);

  // LN + K/V projections (once)
  k_lnproj<<<dim3(B_SZ * N_SZ / 64), 256, 0, stream>>>(
      inputs, g_in, b_in, Wk_bf, Wv_bf, k_g, vT);

  for (int it = 0; it < T_IT; ++it) {
    const int last = (it == T_IT - 1);
    k_qproj<<<B_SZ, 32, 0, stream>>>(slotb, g_s, b_s, Wq_bf, q_g);
    k_logits<<<dim3(N_SZ / 128, B_SZ), 256, 0, stream>>>(q_g, k_g, Lg);
    k_stats<<<B_SZ * K_SL, 256, 0, stream>>>(Lg, stats);
    k_attn<<<dim3(N_SZ / 256, B_SZ), 256, 0, stream>>>(Lg, stats, attnb,
                                                       updts, out_attn, last);
    k_updates<<<dim3(N_SZ / 512, B_SZ), 128, 0, stream>>>(attnb, vT, updts);
    k_gru<<<B_SZ * K_SL, 64, 0, stream>>>(updts, slotb, W_ih, W_hh, b_ih,
                                          b_hh, g_m, b_m, W1, b1, W2, b2,
                                          out_slots, last);
  }
}
